// MambaConvBlock_10969346474335
// MI455X (gfx1250) — matrix operand order repaired and checked by reading, not compiled, not run on hardware
//
#include <hip/hip_runtime.h>
#include <hip/hip_bf16.h>

typedef __attribute__((ext_vector_type(16))) _Float16 v16h;
typedef __attribute__((ext_vector_type(8)))  _Float16 v8h;
typedef __attribute__((ext_vector_type(8)))  float    v8f;
typedef __attribute__((ext_vector_type(4)))  uint32_t u32x4;
typedef __attribute__((ext_vector_type(8)))  uint32_t u32x8;

// ---------------------------------------------------------------------------
// CDNA5 async copy: global -> LDS, 8 bytes per lane, tracked by ASYNCcnt.
// ---------------------------------------------------------------------------
__device__ __forceinline__ void async_b64(uint32_t lds_off, uint32_t goff, uint64_t sbase) {
    asm volatile("global_load_async_to_lds_b64 %0, %1, %2"
                 :: "v"(lds_off), "v"(goff), "s"(sbase) : "memory");
}
__device__ __forceinline__ void wait_async0() {
    asm volatile("s_wait_asynccnt 0" ::: "memory");
}

// ---------------------------------------------------------------------------
// CDNA5 Tensor Data Mover: one scalar-sourced 2-D tile DMA (ISA ch. 7.2/8).
// D# group0: count=1 | lds_addr | global_addr[56:0] | type=2.
// D# group1: data_size=2B, tensor_dim0/1, tile_dim0/1, tensor_dim0_stride.
// Groups 2/3 zero (tile_dim2=0 -> dims >=2 unused). Tracked by TENSORcnt.
// ---------------------------------------------------------------------------
__device__ __forceinline__ void tdm_load_2d(uint32_t lds_addr, uint64_t gaddr,
                                            uint32_t dim0, uint32_t dim1,
                                            uint32_t tile0, uint32_t tile1,
                                            uint32_t stride0) {
    u32x4 g0;
    g0.x = 1u;                                                    // count=1 (user)
    g0.y = lds_addr;                                              // LDS byte addr
    g0.z = (uint32_t)gaddr;                                       // global [31:0]
    g0.w = (uint32_t)((gaddr >> 32) & 0x01FFFFFFu) | (2u << 30);  // [56:32]|type=2
    u32x8 g1;
    g1.s0 = (1u << 16);                                           // data_size=2B
    g1.s1 = (dim0 & 0xFFFFu) << 16;                               // tensor_dim0 lo
    g1.s2 = (dim0 >> 16) | ((dim1 & 0xFFFFu) << 16);              // dim0 hi|dim1 lo
    g1.s3 = (dim1 >> 16) | (tile0 << 16);                         // dim1 hi|tile0
    g1.s4 = tile1;                                                // tile1|tile2=0
    g1.s5 = stride0;                                              // dim0 stride lo
    g1.s6 = 0u;
    g1.s7 = 0u;
    u32x4 gz = {0u, 0u, 0u, 0u};
    asm volatile("tensor_load_to_lds %0, %1, %2, %3"
                 :: "s"(g0), "s"(g1), "s"(gz), "s"(gz) : "memory");
}

// ---------------------------------------------------------------------------
// WMMA fragment loaders (ISA 7.12.2, wave32). LDS row stride 64 halfs.
// ---------------------------------------------------------------------------
__device__ __forceinline__ v16h a_frag(const _Float16* As, int lane, int koff) {
    const int m  = lane & 15;
    const int kb = ((lane >> 4) * 16) + koff;
    v8h lo = *(const v8h*)(As + m * 64 + kb);
    v8h hi = *(const v8h*)(As + m * 64 + kb + 8);
    return __builtin_shufflevector(lo, hi, 0,1,2,3,4,5,6,7,8,9,10,11,12,13,14,15);
}
__device__ __forceinline__ v16h b_frag(const _Float16* Bs, int lane, int koff) {
    const int n  = lane & 15;
    const int kb = ((lane >> 4) * 16) + koff;
    return *(const v16h*)(Bs + n * 64 + kb);
}
__device__ __forceinline__ v8f wmma_f16(v16h a, v16h b, v8f c) {
    return __builtin_amdgcn_wmma_f32_16x16x32_f16(false, a, false, b, (short)0, c, false, false);
}

// Issue one 32Mx64K A tile + 64Nx64K B tile. Constant trip counts: no EXEC
// branching, every thread issues exactly 6 async ops -> uniform ASYNCcnt.
__device__ __forceinline__ void issue_tile(uint32_t aLds, uint32_t bLds,
                                           uint64_t aBase, uint64_t bBase,
                                           int K, int k0, int tid) {
    #pragma unroll
    for (int rnd = 0; rnd < 2; ++rnd) {
        int t = rnd * 256 + tid;
        uint32_t go = (uint32_t)((((t >> 4) * K) + k0 + (t & 15) * 4) * 2);
        async_b64(aLds + t * 8, go, aBase);
    }
    #pragma unroll
    for (int rnd = 0; rnd < 4; ++rnd) {
        int u = rnd * 256 + tid;
        uint32_t go = (uint32_t)((((u >> 4) * K) + k0 + (u & 15) * 4) * 2);
        async_b64(bLds + u * 8, go, bBase);
    }
}

// ---------------------------------------------------------------------------
__global__ void zero_f32(float* p, int n) {
    int i = blockIdx.x * 256 + threadIdx.x;
    if (i < n) p[i] = 0.0f;
}

__global__ void cvt_f16(const float* __restrict__ in, _Float16* __restrict__ out, int n) {
    int i = blockIdx.x * 256 + threadIdx.x;
    if (i < n) out[i] = (_Float16)in[i];
}
// conv weights -> [n][k] with k = (r*3+s)*256 + cin  (K permuted, rs-major)
__global__ void cvt_conv_w(const float* __restrict__ w, _Float16* __restrict__ whc) {
    int i = blockIdx.x * 256 + threadIdx.x;        // 256*2304
    int n = i / 2304, k = i - n * 2304;
    int rs = k >> 8, ci = k & 255;
    whc[i] = (_Float16)w[((size_t)(n * 256 + ci)) * 9 + rs];
}
// x_proj weights (144x512) zero-padded to 192 rows
__global__ void cvt_xproj_pad(const float* __restrict__ w, _Float16* __restrict__ out) {
    int i = blockIdx.x * 256 + threadIdx.x;        // 192*512
    int n = i >> 9, k = i & 511;
    out[i] = (_Float16)((n < 144) ? w[n * 512 + k] : 0.0f);
}

// ---------------------------------------------------------------------------
// Fused conv3x3 + BN + ReLU + 6x6 avg pool (implicit GEMM, WMMA).
// K permuted rs-major: each 64-wide K tile has uniform (r,s).
// Weight tile: one TDM tensor_load_to_lds per tile (wave 0), TENSORcnt.
// im2col A tile: gather-staged (zero padding) while the DMA flies.
// ---------------------------------------------------------------------------
__global__ __launch_bounds__(256)
void conv3x3_bn_relu_pool_wmma(const _Float16* __restrict__ xh,
                               const _Float16* __restrict__ whc,
                               const float* __restrict__ gamma,
                               const float* __restrict__ beta,
                               const float* __restrict__ mean,
                               const float* __restrict__ var,
                               float* __restrict__ pooled) {
    __shared__ __align__(32) _Float16 lds[2048 + 4096];   // A 32x64 | B 64x64
    const uint32_t ldsBase = (uint32_t)(uintptr_t)lds;
    const int tid  = threadIdx.x;
    const int lane = tid & 31;
    const int wave = tid >> 5;
    const int b    = blockIdx.z;
    const int pB   = blockIdx.y * 32;
    const int n0   = blockIdx.x * 64;
    const int wm   = (wave >> 2) * 16;
    const int wn   = (wave & 3) * 16;

    v8f acc = {};
    for (int k0 = 0; k0 < 2304; k0 += 64) {
        // TDM: 64 rows x 128B weight tile, row stride 4608B -> contiguous LDS
        if (wave == 0)
            tdm_load_2d(ldsBase + 4096,
                        (uint64_t)(uintptr_t)(whc + (size_t)n0 * 2304 + k0),
                        /*dim0*/2304u, /*dim1*/64u, /*tile0*/64u, /*tile1*/64u,
                        /*stride0*/2304u);
        // gather: im2col A tile; (r,s) uniform over this K tile
        const int rs = k0 >> 8;
        const int r  = rs / 3, s = rs - r * 3;
        const int cb = k0 & 255;
        #pragma unroll
        for (int rnd = 0; rnd < 8; ++rnd) {
            int i  = rnd * 256 + tid;
            int m  = i >> 6, kk = i & 63;
            int ci = cb + kk;
            int p  = pB + m;
            int hh = p / 96 + r - 1;
            int ww = p - (p / 96) * 96 + s - 1;
            _Float16 v = (_Float16)0.0f;
            if (hh >= 0 && hh < 96 && ww >= 0 && ww < 96)
                v = xh[(((size_t)b * 256 + ci) * 96 + hh) * 96 + ww];
            lds[m * 64 + kk] = v;
        }
        if (wave == 0)
            __builtin_amdgcn_s_wait_tensorcnt((short)0);
        __syncthreads();
        const _Float16* Asb = lds + wm * 64;
        const _Float16* Bsb = lds + 2048 + wn * 64;
        acc = wmma_f16(a_frag(Asb, lane, 0),  b_frag(Bsb, lane, 0),  acc);
        acc = wmma_f16(a_frag(Asb, lane, 32), b_frag(Bsb, lane, 32), acc);
        __syncthreads();
    }

    // BN + ReLU + pooled accumulation (C/D: lanes 0-15 M=r, 16-31 M=r+8)
    const int ch  = n0 + wn + (lane & 15);
    const float inv = gamma[ch] * rsqrtf(var[ch] + 1e-5f);
    const float bb2 = beta[ch] - mean[ch] * inv;
    const int mB  = pB + wm + (lane >> 4) * 8;
    #pragma unroll
    for (int rr = 0; rr < 8; ++rr) {
        int p = mB + rr;
        float y = fmaxf(acc[rr] * inv + bb2, 0.0f);
        int h = p / 96, ww = p - h * 96;
        int cell = (h / 6) * 16 + (ww / 6);
        atomicAdd(&pooled[((size_t)b * 256 + ch) * 256 + cell], y * (1.0f / 36.0f));
    }
}

// ---------------------------------------------------------------------------
__global__ void build_tokens(const float* __restrict__ pooled,
                             _Float16* __restrict__ tokens) {
    int i = blockIdx.x * 256 + threadIdx.x;   // 524288
    int c = i & 255;
    int bl = i >> 8;
    int b = bl >> 8, l = bl & 255;
    tokens[i] = (_Float16)pooled[((size_t)b * 256 + c) * 256 + l];
}

// ---------------------------------------------------------------------------
// Generic async double-buffered WMMA GEMM: C[M,Nstore] = A[M,K] * B[N,K]^T
// Tile k+1 issued (ASYNCcnt) while tile k computes; 2 WMMA / wave / tile.
// ---------------------------------------------------------------------------
__global__ __launch_bounds__(256)
void gemm_wmma_async(const _Float16* __restrict__ A, const _Float16* __restrict__ B,
                     float* __restrict__ C, int Nstore, int K) {
    __shared__ __align__(32) _Float16 lds[2 * 2048 + 2 * 4096];  // A0 A1 B0 B1
    const uint32_t ldsBase = (uint32_t)(uintptr_t)lds;
    const int tid  = threadIdx.x;
    const int lane = tid & 31;
    const int wave = tid >> 5;
    const int m0 = blockIdx.y * 32;
    const int n0 = blockIdx.x * 64;
    const int wm = (wave >> 2) * 16;
    const int wn = (wave & 3) * 16;
    const uint64_t aBase = (uint64_t)(uintptr_t)(A + (size_t)m0 * K);
    const uint64_t bBase = (uint64_t)(uintptr_t)(B + (size_t)n0 * K);
    const int nIter = K >> 6;

    v8f acc = {};
    issue_tile(ldsBase, ldsBase + 8192, aBase, bBase, K, 0, tid);
    for (int it = 0; it < nIter; ++it) {
        const int cur = it & 1;
        wait_async0();          // drain this tile's async writes to LDS
        __syncthreads();        // publish across all 8 waves
        if (it + 1 < nIter)     // overlap next tile's loads with compute
            issue_tile(ldsBase + (cur ^ 1) * 4096,
                       ldsBase + 8192 + (cur ^ 1) * 8192,
                       aBase, bBase, K, (it + 1) << 6, tid);
        const _Float16* Asb = lds + cur * 2048 + wm * 64;
        const _Float16* Bsb = lds + 4096 + cur * 4096 + wn * 64;
        acc = wmma_f16(a_frag(Asb, lane, 0),  b_frag(Bsb, lane, 0),  acc);
        acc = wmma_f16(a_frag(Asb, lane, 32), b_frag(Bsb, lane, 32), acc);
        __syncthreads();        // done reading buf[cur] before its reuse
    }
    const int n_g = n0 + wn + (lane & 15);
    const int m_b = m0 + wm + (lane >> 4) * 8;
    if (n_g < Nstore) {
        #pragma unroll
        for (int rr = 0; rr < 8; ++rr)
            C[(size_t)(m_b + rr) * Nstore + n_g] = acc[rr];
    }
}

// ---------------------------------------------------------------------------
__global__ void conv1d_silu(const float* __restrict__ xz,
                            const float* __restrict__ w,
                            const float* __restrict__ bias,
                            float* __restrict__ xc,
                            _Float16* __restrict__ xch) {
    int id = blockIdx.x * 256 + threadIdx.x;  // 2048*512
    int d = id & 511;
    int bl = id >> 9;
    int b = bl >> 8, l = bl & 255;
    float acc = bias[d];
    #pragma unroll
    for (int j = 0; j < 4; ++j) {
        int ll = l - 3 + j;
        if (ll >= 0)
            acc += w[d * 4 + j] * xz[((size_t)(b * 256 + ll)) * 1024 + d];
    }
    float s = acc / (1.0f + __expf(-acc));
    xc[(size_t)bl * 512 + d]  = s;
    xch[(size_t)bl * 512 + d] = (_Float16)s;
}

// ---------------------------------------------------------------------------
__global__ void dtproj_softplus(const float* __restrict__ xdbl,
                                const float* __restrict__ w,
                                const float* __restrict__ bias,
                                float* __restrict__ dt) {
    int id = blockIdx.x * 256 + threadIdx.x;  // 2048*512
    int d = id & 511;
    int bl = id >> 9;
    float acc = bias[d];
    #pragma unroll
    for (int j = 0; j < 16; ++j)
        acc += xdbl[(size_t)bl * 144 + j] * w[d * 16 + j];
    dt[(size_t)bl * 512 + d] = (acc > 20.0f) ? acc : log1pf(__expf(acc));
}

// ---------------------------------------------------------------------------
__global__ __launch_bounds__(256)
void scan_kernel(const float* __restrict__ dt, const float* __restrict__ xc,
                 const float* __restrict__ xdbl, const float* __restrict__ xz,
                 const float* __restrict__ A_log, const float* __restrict__ Dp,
                 _Float16* __restrict__ ysh) {
    const int lane = threadIdx.x & 31;
    const int wave = threadIdx.x >> 5;
    const int idx  = blockIdx.x * 8 + wave;   // 0..4095
    const int b = idx >> 9;
    const int d = idx & 511;

    const float A0 = -__expf(A_log[d * 64 + lane]);
    const float A1 = -__expf(A_log[d * 64 + 32 + lane]);
    const float Dv = Dp[d];

    float h0 = 0.0f, h1 = 0.0f;
    for (int l = 0; l < 256; ++l) {
        const size_t bl = (size_t)b * 256 + l;
        float dtv = dt[bl * 512 + d];
        float xcv = xc[bl * 512 + d];
        float B0 = xdbl[bl * 144 + 16 + lane];
        float B1 = xdbl[bl * 144 + 48 + lane];
        float C0 = xdbl[bl * 144 + 80 + lane];
        float C1 = xdbl[bl * 144 + 112 + lane];
        float du = dtv * xcv;
        h0 = __expf(dtv * A0) * h0 + du * B0;
        h1 = __expf(dtv * A1) * h1 + du * B1;
        float y = h0 * C0 + h1 * C1;
        #pragma unroll
        for (int off = 16; off > 0; off >>= 1)
            y += __shfl_xor(y, off, 32);
        if (lane == 0) {
            float zv = xz[bl * 1024 + 512 + d];
            float gate = zv / (1.0f + __expf(-zv));
            ysh[bl * 512 + d] = (_Float16)((y + xcv * Dv) * gate);
        }
    }
}

// ---------------------------------------------------------------------------
__global__ void upsample_residual_relu(const float* __restrict__ ym,
                                       const float* __restrict__ x,
                                       float* __restrict__ out) {
    size_t id = (size_t)blockIdx.x * 256 + threadIdx.x;  // 8*256*96*96
    int b = (int)(id / (256u * 9216u));
    int rem = (int)(id - (size_t)b * 256 * 9216);
    int c = rem / 9216;
    int p = rem - c * 9216;
    int h = p / 96, w = p - h * 96;

    float sy = (h + 0.5f) * (1.0f / 6.0f) - 0.5f;
    float sx = (w + 0.5f) * (1.0f / 6.0f) - 0.5f;
    int y0 = (int)floorf(sy), x0 = (int)floorf(sx);
    float fy = sy - y0, fx = sx - x0;
    int y1 = min(max(y0 + 1, 0), 15), x1 = min(max(x0 + 1, 0), 15);
    y0 = min(max(y0, 0), 15);
    x0 = min(max(x0, 0), 15);

    const size_t base = (size_t)b * 256;
    float v00 = ym[(base + y0 * 16 + x0) * 256 + c];
    float v01 = ym[(base + y0 * 16 + x1) * 256 + c];
    float v10 = ym[(base + y1 * 16 + x0) * 256 + c];
    float v11 = ym[(base + y1 * 16 + x1) * 256 + c];
    float v0 = v00 + (v01 - v00) * fx;
    float v1 = v10 + (v11 - v10) * fx;
    float v = v0 + (v1 - v0) * fy;

    out[id] = fmaxf(v + x[id], 0.0f);
}

// ---------------------------------------------------------------------------
extern "C" void kernel_launch(void* const* d_in, const int* in_sizes, int n_in,
                              void* d_out, int out_size, void* d_ws, size_t ws_size,
                              hipStream_t stream) {
    (void)in_sizes; (void)n_in; (void)out_size; (void)ws_size;
    const float* x          = (const float*)d_in[0];
    const float* conv_w     = (const float*)d_in[1];
    const float* bn_gamma   = (const float*)d_in[2];
    const float* bn_beta    = (const float*)d_in[3];
    const float* bn_mean    = (const float*)d_in[4];
    const float* bn_var     = (const float*)d_in[5];
    const float* in_proj_w  = (const float*)d_in[6];
    const float* conv1d_w   = (const float*)d_in[7];
    const float* conv1d_b   = (const float*)d_in[8];
    const float* x_proj_w   = (const float*)d_in[9];
    const float* dt_proj_w  = (const float*)d_in[10];
    const float* dt_proj_b  = (const float*)d_in[11];
    const float* A_log      = (const float*)d_in[12];
    const float* Dp         = (const float*)d_in[13];
    const float* out_proj_w = (const float*)d_in[14];
    float* out = (float*)d_out;

    // f32 scratch
    float* ws     = (float*)d_ws;
    float* pooled = ws; ws += 8 * 256 * 256;       // 524288
    float* xz     = ws; ws += 2048 * 1024;         // 2097152
    float* xc     = ws; ws += 2048 * 512;          // 1048576
    float* xdbl   = ws; ws += 2048 * 144;          // 294912
    float* dt     = ws; ws += 2048 * 512;          // 1048576
    float* ym     = ws; ws += 2048 * 256;          // 524288
    // f16 scratch
    _Float16* hs   = (_Float16*)ws;
    _Float16* xh   = hs; hs += 8 * 256 * 96 * 96;  // 18874368
    _Float16* whc  = hs; hs += 256 * 2304;         // 589824
    _Float16* wip  = hs; hs += 1024 * 256;         // 262144
    _Float16* wxp  = hs; hs += 192 * 512;          // 98304 (padded 144->192)
    _Float16* wop  = hs; hs += 256 * 512;          // 131072
    _Float16* tokh = hs; hs += 2048 * 256;         // 524288
    _Float16* xch  = hs; hs += 2048 * 512;         // 1048576
    _Float16* ysh  = hs; hs += 2048 * 512;         // 1048576

    zero_f32<<<2048, 256, 0, stream>>>(pooled, 524288);
    cvt_f16<<<73728, 256, 0, stream>>>(x, xh, 18874368);
    cvt_conv_w<<<2304, 256, 0, stream>>>(conv_w, whc);
    cvt_f16<<<1024, 256, 0, stream>>>(in_proj_w, wip, 262144);
    cvt_xproj_pad<<<384, 256, 0, stream>>>(x_proj_w, wxp);
    cvt_f16<<<512, 256, 0, stream>>>(out_proj_w, wop, 131072);

    conv3x3_bn_relu_pool_wmma<<<dim3(4, 288, 8), 256, 0, stream>>>(
        xh, whc, bn_gamma, bn_beta, bn_mean, bn_var, pooled);
    build_tokens<<<2048, 256, 0, stream>>>(pooled, tokh);
    // in_proj: (2048,256) x (1024,256)^T
    gemm_wmma_async<<<dim3(16, 64), 256, 0, stream>>>(tokh, wip, xz, 1024, 256);
    conv1d_silu<<<4096, 256, 0, stream>>>(xz, conv1d_w, conv1d_b, xc, xch);
    // x_proj: (2048,512) x (144,512)^T  (B padded to 192 rows)
    gemm_wmma_async<<<dim3(3, 64), 256, 0, stream>>>(xch, wxp, xdbl, 144, 512);
    dtproj_softplus<<<4096, 256, 0, stream>>>(xdbl, dt_proj_w, dt_proj_b, dt);
    scan_kernel<<<512, 256, 0, stream>>>(dt, xc, xdbl, xz, A_log, Dp, ysh);
    // out_proj: (2048,512) x (256,512)^T
    gemm_wmma_async<<<dim3(4, 64), 256, 0, stream>>>(ysh, wop, ym, 256, 512);
    upsample_residual_relu<<<73728, 256, 0, stream>>>(ym, x, out);
}
